// TGAE_29111288332322
// MI455X (gfx1250) — matrix operand order repaired and checked by reading, not compiled, not run on hardware
//
#include <hip/hip_runtime.h>
#include <hip/hip_bf16.h>

// ---------------------------------------------------------------------------
// Types for CDNA5 WMMA (wave32)
// ---------------------------------------------------------------------------
typedef __bf16 bf16_t;
typedef __attribute__((ext_vector_type(16))) bf16_t v16bf;
typedef __attribute__((ext_vector_type(8)))  bf16_t v8bf;
typedef __attribute__((ext_vector_type(8)))  float  v8f;

#define N_NODES 65536
#define F_IN    128
#define HDIM    256
#define LDIM    64
#define E_EDGES 2097152
#define EP_PRED 1048576
#define CHUNK   16384   // row chunk for KAN GEMMs (divides N_NODES and EP_PRED)

// ---------------------------------------------------------------------------
// Device helpers
// ---------------------------------------------------------------------------
__device__ __forceinline__ bf16_t f2bf(float f) {
  union { float f; unsigned u; } v; v.f = f;
  unsigned r = v.u + 0x7FFFu + ((v.u >> 16) & 1u);   // RNE
  unsigned short s = (unsigned short)(r >> 16);
  return __builtin_bit_cast(bf16_t, s);
}

__device__ __forceinline__ float silu_f(float x) {
  return x * (1.0f / (1.0f + __expf(-x)));
}

// Cox-de Boor cubic B-spline bases on uniform knots t[j] = -2.2 + 0.4*j
// (GS=5, K=3 -> 8 bases). Matches reference b_splines().
__device__ __forceinline__ void bspline8(float x, float* __restrict__ out) {
  const float h = 0.4f;
  float b[11];
#pragma unroll
  for (int j = 0; j < 11; ++j) {
    float t0 = -2.2f + h * j;
    b[j] = (x >= t0 && x < t0 + h) ? 1.0f : 0.0f;
  }
#pragma unroll
  for (int p = 1; p <= 3; ++p) {
    float inv = 1.0f / (h * (float)p);
#pragma unroll
    for (int j = 0; j + p < 11; ++j) {
      float tj  = -2.2f + h * j;
      float tjp = tj + h * (float)p;
      b[j] = (x - tj) * inv * b[j] + (tjp + h - x) * inv * b[j + 1];
    }
  }
#pragma unroll
  for (int k = 0; k < 8; ++k) out[k] = b[k];
}

// ---------------------------------------------------------------------------
// Small utility kernels
// ---------------------------------------------------------------------------
__global__ void fill_kernel(float* p, float v, int n) {
  int i = blockIdx.x * blockDim.x + threadIdx.x;
  if (i < n) p[i] = v;
}

__global__ void edge_deg_kernel(const int* __restrict__ dst, float* __restrict__ deg, int e) {
  int i = blockIdx.x * blockDim.x + threadIdx.x;
  if (i < e) atomicAdd(&deg[dst[i]], 1.0f);
}

__global__ void dinv_kernel(const float* __restrict__ deg, float* __restrict__ dinv, int n) {
  int i = blockIdx.x * blockDim.x + threadIdx.x;
  if (i < n) dinv[i] = deg[i] > 0.0f ? __frsqrt_rn(deg[i]) : 0.0f;
}

__global__ void cvt_bf16_kernel(const float* __restrict__ src, bf16_t* __restrict__ dst, int n) {
  int i = blockIdx.x * blockDim.x + threadIdx.x;
  if (i < n) dst[i] = f2bf(src[i]);
}

// out[i,f] = xw[i,f] * dinv[i]^2     (self-loop term; H == 256)
__global__ void selfloop_kernel(const float* __restrict__ xw, const float* __restrict__ dinv,
                                float* __restrict__ out, int n) {
  int idx = blockIdx.x * blockDim.x + threadIdx.x;
  if (idx < n * HDIM) {
    int i = idx >> 8;
    float d = dinv[i];
    out[idx] = xw[idx] * d * d;
  }
}

// scatter-add normalized messages; one thread = one (edge, 4-feature group)
__global__ void aggregate_kernel(const float* __restrict__ xw, const int* __restrict__ src,
                                 const int* __restrict__ dst, const float* __restrict__ dinv,
                                 float* __restrict__ out, int e) {
  long long idx = (long long)blockIdx.x * blockDim.x + threadIdx.x;
  if (idx >= (long long)e * (HDIM / 4)) return;
  int ei = (int)(idx >> 6);
  int q  = (int)(idx & 63);
  int s = src[ei], d = dst[ei];
  float c = dinv[s] * dinv[d];
  const float4 v = *(const float4*)&xw[(size_t)s * HDIM + q * 4];
  float* o = &out[(size_t)d * HDIM + q * 4];
  atomicAdd(o + 0, v.x * c);
  atomicAdd(o + 1, v.y * c);
  atomicAdd(o + 2, v.z * c);
  atomicAdd(o + 3, v.w * c);
}

__global__ void bias_relu_kernel(float* __restrict__ out, const float* __restrict__ b, int n) {
  int idx = blockIdx.x * blockDim.x + threadIdx.x;
  if (idx < n * HDIM) out[idx] = fmaxf(out[idx] + b[idx & 255], 0.0f);
}

// ef[r,l] = z[e0[base+r],l] * z[e1[base+r],l]   (L == 64)
__global__ void gather_mul_kernel(const int* __restrict__ e0, const int* __restrict__ e1,
                                  const float* __restrict__ z, float* __restrict__ ef,
                                  int base, int rows) {
  int idx = blockIdx.x * blockDim.x + threadIdx.x;
  if (idx >= rows * LDIM) return;
  int r = idx >> 6, l = idx & 63;
  int g = base + r;
  ef[idx] = z[(size_t)e0[g] * LDIM + l] * z[(size_t)e1[g] * LDIM + l];
}

// Expand [rows, in] f32 -> [rows, 9*in] bf16: [silu | 8 spline bases per input]
__global__ void expand_kernel(const float* __restrict__ src, bf16_t* __restrict__ dst,
                              int rows, int in) {
  int idx = blockIdx.x * blockDim.x + threadIdx.x;
  if (idx >= rows * in) return;
  int r = idx / in, i = idx - r * in;
  float x = src[idx];
  size_t base = (size_t)r * (9 * in);
  dst[base + i] = f2bf(silu_f(x));
  float b[8];
  bspline8(x, b);
  alignas(16) bf16_t tmp[8];
#pragma unroll
  for (int k = 0; k < 8; ++k) tmp[k] = f2bf(b[k]);
  *(uint4*)&dst[base + in + (size_t)i * 8] = *(const uint4*)tmp;  // 16B aligned
}

// Pack KAN weights into bf16 [9*in, out] (K-major): row i -> bw, rows in+i*8+k -> sw*sc
__global__ void pack_kan_kernel(const float* __restrict__ bw, const float* __restrict__ sw,
                                const float* __restrict__ sc, bf16_t* __restrict__ wp,
                                int in, int out) {
  int idx = blockIdx.x * blockDim.x + threadIdx.x;
  if (idx >= in * out) return;
  int o = idx / in, i = idx - o * in;
  wp[(size_t)i * out + o] = f2bf(bw[(size_t)o * in + i]);
  float s = sc[(size_t)o * in + i];
#pragma unroll
  for (int k = 0; k < 8; ++k)
    wp[(size_t)(in + i * 8 + k) * out + o] = f2bf(sw[((size_t)o * in + i) * 8 + k] * s);
}

// ---------------------------------------------------------------------------
// WMMA GEMM: C[M,N] f32 = A[M,K] bf16 (row-major) x B[K,N] bf16 (row-major)
// Block: 256 threads (8 waves). Block tile 256(M) x 64(N), K-step 32,
// double-buffered LDS. A tiles staged with GLOBAL_LOAD_ASYNC_TO_LDS_B128
// (ASYNCcnt), overlapped with WMMA work; B tiles staged transposed via VGPRs.
// Each wave: 32 rows x 64 cols = 8 x v_wmma_f32_16x16x32_bf16 per K-step,
// with B-fragment reuse across the two A fragments.
// ---------------------------------------------------------------------------
__global__ __launch_bounds__(256)
void wmma_gemm_kernel(const bf16_t* __restrict__ A, const bf16_t* __restrict__ B,
                      float* __restrict__ C, int M, int N, int K,
                      const float* __restrict__ bias, int relu) {
  __shared__ alignas(32) bf16_t As[2][256 * 32];  // row-major [m][k]
  __shared__ alignas(32) bf16_t Bs[2][64 * 32];   // transposed  [n][k]

  const int t = threadIdx.x;
  const int wave = t >> 5;
  const int lane = t & 31;
  const int m0 = blockIdx.y * 256;
  const int n0 = blockIdx.x * 64;

  v8f acc[2][4] = {};

  // lane mapping: contiguous k = 16*(lane/16) + s, same for A and B (not the ISA table order)
  const int mrow = lane & 15;
  const int kbA  = (lane & 16) ? 16 : 0;   // A: lanes 16-31 hold K=16..31
  const int kbB  = (lane & 16) ? 16 : 0;   // B: lanes 16-31 hold K=16..31
  const int ncol = lane & 15;

  const int bk = t >> 3;                   // B stage: 32 k x (8 x 8 n)
  const int bn = (t & 7) * 8;

  // LDS byte offsets for async-DMA destinations (thread t stages row t: 64B)
  const unsigned ldsA[2] = { (unsigned)(uintptr_t)&As[0][0] + (unsigned)t * 64u,
                             (unsigned)(uintptr_t)&As[1][0] + (unsigned)t * 64u };

  auto stage = [&](int k0, int buf) {
    // A: 4 async 16B transfers per thread, written directly into LDS
    unsigned long long g =
        (unsigned long long)(uintptr_t)(A + (size_t)(m0 + t) * K + k0);
#pragma unroll
    for (int q = 0; q < 4; ++q) {
      asm volatile("global_load_async_to_lds_b128 %0, %1, off"
                   :: "v"(ldsA[buf] + (unsigned)(q * 16)),
                      "v"(g + (unsigned long long)(q * 16))
                   : "memory");
    }
    // B: vector global load, transposed scatter into LDS
    v8bf bv = *(const v8bf*)&B[(size_t)(k0 + bk) * N + n0 + bn];
    bf16_t* d = &Bs[buf][0];
#pragma unroll
    for (int q = 0; q < 8; ++q) d[(bn + q) * 32 + bk] = bv[q];
  };

  const int T = K >> 5;
  stage(0, 0);

  for (int ti = 0; ti < T; ++ti) {
    asm volatile("s_wait_asynccnt 0x0" ::: "memory");
    __syncthreads();                       // tile ti staged & visible to all waves
    if (ti + 1 < T) stage((ti + 1) * 32, (ti + 1) & 1);  // overlap DMA with math

    const int b = ti & 1;
    const bf16_t* arow0 = &As[b][(wave * 32 + mrow) * 32];
    const bf16_t* arow1 = arow0 + 16 * 32;
    v8bf a0l = *(const v8bf*)(arow0 + kbA);
    v8bf a0h = *(const v8bf*)(arow0 + kbA + 8);
    v8bf a1l = *(const v8bf*)(arow1 + kbA);
    v8bf a1h = *(const v8bf*)(arow1 + kbA + 8);
    v16bf af0 = __builtin_shufflevector(a0l, a0h,
        0, 1, 2, 3, 4, 5, 6, 7, 8, 9, 10, 11, 12, 13, 14, 15);
    v16bf af1 = __builtin_shufflevector(a1l, a1h,
        0, 1, 2, 3, 4, 5, 6, 7, 8, 9, 10, 11, 12, 13, 14, 15);
#pragma unroll
    for (int j = 0; j < 4; ++j) {
      v16bf bf = *(const v16bf*)&Bs[b][(j * 16 + ncol) * 32 + kbB];
      acc[0][j] = __builtin_amdgcn_wmma_f32_16x16x32_bf16(
          false, af0, false, bf, (short)0, acc[0][j], false, false);
      acc[1][j] = __builtin_amdgcn_wmma_f32_16x16x32_bf16(
          false, af1, false, bf, (short)0, acc[1][j], false, false);
    }
  }

  // Epilogue per documented C layout: VGPR r -> M = r + 8*(lane>=16), N = lane&15
  const int moff = (lane & 16) ? 8 : 0;
#pragma unroll
  for (int i = 0; i < 2; ++i) {
#pragma unroll
    for (int j = 0; j < 4; ++j) {
      int col = n0 + j * 16 + ncol;
      float bv = bias ? bias[col] : 0.0f;
#pragma unroll
      for (int r = 0; r < 8; ++r) {
        int row = m0 + wave * 32 + i * 16 + moff + r;
        float v = acc[i][j][r] + bv;
        if (relu) v = fmaxf(v, 0.0f);
        C[(size_t)row * N + col] = v;
      }
    }
  }
}

// ---------------------------------------------------------------------------
// KAN layer with out==1 (ek2 / dk2): one wave per row, lane-parallel over `in`,
// wave reduction, then sigmoid (ACT=0) or relu (ACT=1).
// ---------------------------------------------------------------------------
template <int ACT>
__global__ void kan_out1_kernel(const float* __restrict__ A, int rows, int in,
                                const float* __restrict__ bw, const float* __restrict__ sw,
                                const float* __restrict__ sc, float* __restrict__ out) {
  int wave = threadIdx.x >> 5;
  int lane = threadIdx.x & 31;
  int row = blockIdx.x * 8 + wave;
  if (row >= rows) return;
  const float* a = A + (size_t)row * in;
  float acc = 0.0f;
  for (int i = lane; i < in; i += 32) {
    float x = a[i];
    float b[8];
    bspline8(x, b);
    float s = sc[i];
    float part = silu_f(x) * bw[i];
#pragma unroll
    for (int k = 0; k < 8; ++k) part += b[k] * sw[i * 8 + k] * s;
    acc += part;
  }
#pragma unroll
  for (int off = 16; off; off >>= 1) acc += __shfl_xor(acc, off);
  if (lane == 0)
    out[row] = (ACT == 0) ? (1.0f / (1.0f + __expf(-acc))) : fmaxf(acc, 0.0f);
}

// ---------------------------------------------------------------------------
// Host orchestration
// ---------------------------------------------------------------------------
static inline void launch_gemm(const bf16_t* A, const bf16_t* B, float* C,
                               int M, int N, int K, const float* bias, int relu,
                               hipStream_t s) {
  dim3 grid(N / 64, M / 256);
  wmma_gemm_kernel<<<grid, 256, 0, s>>>(A, B, C, M, N, K, bias, relu);
}

extern "C" void kernel_launch(void* const* d_in, const int* in_sizes, int n_in,
                              void* d_out, int out_size, void* d_ws, size_t ws_size,
                              hipStream_t stream) {
  (void)in_sizes; (void)n_in; (void)out_size; (void)ws_size;

  const float* x        = (const float*)d_in[0];
  const int*   ei       = (const int*)d_in[1];   // [2,E] : src row 0, dst row 1
  const int*   ep       = (const int*)d_in[2];   // [2,EP]
  const float* gcn1_w   = (const float*)d_in[3];
  const float* gcn1_b   = (const float*)d_in[4];
  const float* gcn2_w   = (const float*)d_in[5];
  const float* gcn2_b   = (const float*)d_in[6];
  const float* dec_w    = (const float*)d_in[7];
  const float* dec_b    = (const float*)d_in[8];
  const float* enc_bw   = (const float*)d_in[9];
  const float* enc_sw   = (const float*)d_in[10];
  const float* enc_sc   = (const float*)d_in[11];
  const float* feat_bw  = (const float*)d_in[12];
  const float* feat_sw  = (const float*)d_in[13];
  const float* feat_sc  = (const float*)d_in[14];
  const float* ek1_bw   = (const float*)d_in[15];
  const float* ek1_sw   = (const float*)d_in[16];
  const float* ek1_sc   = (const float*)d_in[17];
  const float* ek2_bw   = (const float*)d_in[18];
  const float* ek2_sw   = (const float*)d_in[19];
  const float* ek2_sc   = (const float*)d_in[20];
  const float* dk1_bw   = (const float*)d_in[21];
  const float* dk1_sw   = (const float*)d_in[22];
  const float* dk1_sc   = (const float*)d_in[23];
  const float* dk2_bw   = (const float*)d_in[24];
  const float* dk2_sw   = (const float*)d_in[25];
  const float* dk2_sc   = (const float*)d_in[26];

  float* out = (float*)d_out;
  float* recon_out  = out;                                             // [N,128]
  float* edge_out   = out + (size_t)N_NODES * F_IN;                    // [EP]
  float* deg_out    = edge_out + EP_PRED;                              // [N]
  float* z          = deg_out + N_NODES;                               // [N,64]

  // workspace carving (256B aligned)
  size_t off = 0;
  char* ws = (char*)d_ws;
  auto carve = [&](size_t bytes) -> void* {
    void* p = ws + off;
    off += (bytes + 255) & ~(size_t)255;
    return p;
  };
  float*  deg    = (float*)carve((size_t)N_NODES * 4);
  float*  dinv   = (float*)carve((size_t)N_NODES * 4);
  float*  bufA   = (float*)carve((size_t)N_NODES * HDIM * 4);  // xw results
  float*  bufB   = (float*)carve((size_t)N_NODES * HDIM * 4);  // aggregated h
  bf16_t* stage  = (bf16_t*)carve((size_t)N_NODES * HDIM * 2); // bf16 GEMM A input
  bf16_t* expb   = (bf16_t*)carve((size_t)CHUNK * (9 * HDIM) * 2);
  float*  ehb    = (float*)carve((size_t)CHUNK * HDIM * 4);
  float*  efb    = (float*)carve((size_t)CHUNK * LDIM * 4);
  bf16_t* w_gcn1 = (bf16_t*)carve((size_t)F_IN * HDIM * 2);
  bf16_t* w_gcn2 = (bf16_t*)carve((size_t)HDIM * HDIM * 2);
  bf16_t* w_dec  = (bf16_t*)carve((size_t)LDIM * HDIM * 2);
  bf16_t* wp_enc = (bf16_t*)carve((size_t)(9 * HDIM) * LDIM * 2);
  bf16_t* wp_feat= (bf16_t*)carve((size_t)(9 * HDIM) * F_IN * 2);
  bf16_t* wp_ek1 = (bf16_t*)carve((size_t)(9 * LDIM) * HDIM * 2);
  bf16_t* wp_dk1 = (bf16_t*)carve((size_t)(9 * LDIM) * HDIM * 2);

  const int* e_src = ei;
  const int* e_dst = ei + E_EDGES;
  const int* p0    = ep;
  const int* p1    = ep + EP_PRED;

  auto blocks = [](long long n, int bs) { return (unsigned)((n + bs - 1) / bs); };

  // ---- weight prep ----
  cvt_bf16_kernel<<<blocks(F_IN * HDIM, 256), 256, 0, stream>>>(gcn1_w, w_gcn1, F_IN * HDIM);
  cvt_bf16_kernel<<<blocks(HDIM * HDIM, 256), 256, 0, stream>>>(gcn2_w, w_gcn2, HDIM * HDIM);
  cvt_bf16_kernel<<<blocks(LDIM * HDIM, 256), 256, 0, stream>>>(dec_w, w_dec, LDIM * HDIM);
  pack_kan_kernel<<<blocks(HDIM * LDIM, 256), 256, 0, stream>>>(enc_bw, enc_sw, enc_sc, wp_enc, HDIM, LDIM);
  pack_kan_kernel<<<blocks(HDIM * F_IN, 256), 256, 0, stream>>>(feat_bw, feat_sw, feat_sc, wp_feat, HDIM, F_IN);
  pack_kan_kernel<<<blocks(LDIM * HDIM, 256), 256, 0, stream>>>(ek1_bw, ek1_sw, ek1_sc, wp_ek1, LDIM, HDIM);
  pack_kan_kernel<<<blocks(LDIM * HDIM, 256), 256, 0, stream>>>(dk1_bw, dk1_sw, dk1_sc, wp_dk1, LDIM, HDIM);

  // ---- symmetric-normalization degrees (with self loops) ----
  fill_kernel<<<blocks(N_NODES, 256), 256, 0, stream>>>(deg, 1.0f, N_NODES);
  edge_deg_kernel<<<blocks(E_EDGES, 256), 256, 0, stream>>>(e_dst, deg, E_EDGES);
  dinv_kernel<<<blocks(N_NODES, 256), 256, 0, stream>>>(deg, dinv, N_NODES);

  // ---- GCN layer 1: h1 = relu(agg(x @ W1) + b1) ----
  cvt_bf16_kernel<<<blocks((long long)N_NODES * F_IN, 256), 256, 0, stream>>>(x, stage, N_NODES * F_IN);
  launch_gemm(stage, w_gcn1, bufA, N_NODES, HDIM, F_IN, nullptr, 0, stream);
  selfloop_kernel<<<blocks((long long)N_NODES * HDIM, 256), 256, 0, stream>>>(bufA, dinv, bufB, N_NODES);
  aggregate_kernel<<<blocks((long long)E_EDGES * (HDIM / 4), 256), 256, 0, stream>>>(bufA, e_src, e_dst, dinv, bufB, E_EDGES);
  bias_relu_kernel<<<blocks((long long)N_NODES * HDIM, 256), 256, 0, stream>>>(bufB, gcn1_b, N_NODES);

  // ---- GCN layer 2: h2 = relu(agg(h1 @ W2) + b2) ----
  cvt_bf16_kernel<<<blocks((long long)N_NODES * HDIM, 256), 256, 0, stream>>>(bufB, stage, N_NODES * HDIM);
  launch_gemm(stage, w_gcn2, bufA, N_NODES, HDIM, HDIM, nullptr, 0, stream);
  selfloop_kernel<<<blocks((long long)N_NODES * HDIM, 256), 256, 0, stream>>>(bufA, dinv, bufB, N_NODES);
  aggregate_kernel<<<blocks((long long)E_EDGES * (HDIM / 4), 256), 256, 0, stream>>>(bufA, e_src, e_dst, dinv, bufB, E_EDGES);
  bias_relu_kernel<<<blocks((long long)N_NODES * HDIM, 256), 256, 0, stream>>>(bufB, gcn2_b, N_NODES);

  // ---- encoder KAN: z = kan(h2) [N,64] ----
  for (int c = 0; c < N_NODES / CHUNK; ++c) {
    expand_kernel<<<blocks((long long)CHUNK * HDIM, 256), 256, 0, stream>>>(
        bufB + (size_t)c * CHUNK * HDIM, expb, CHUNK, HDIM);
    launch_gemm(expb, wp_enc, z + (size_t)c * CHUNK * LDIM, CHUNK, LDIM, 9 * HDIM, nullptr, 0, stream);
  }

  // ---- feature decoder GCN: hd = relu(agg(z @ Wdec) + bdec) ----
  cvt_bf16_kernel<<<blocks((long long)N_NODES * LDIM, 256), 256, 0, stream>>>(z, stage, N_NODES * LDIM);
  launch_gemm(stage, w_dec, bufA, N_NODES, HDIM, LDIM, nullptr, 0, stream);
  selfloop_kernel<<<blocks((long long)N_NODES * HDIM, 256), 256, 0, stream>>>(bufA, dinv, bufB, N_NODES);
  aggregate_kernel<<<blocks((long long)E_EDGES * (HDIM / 4), 256), 256, 0, stream>>>(bufA, e_src, e_dst, dinv, bufB, E_EDGES);
  bias_relu_kernel<<<blocks((long long)N_NODES * HDIM, 256), 256, 0, stream>>>(bufB, dec_b, N_NODES);

  // ---- feature decoder KAN: reconstructed_x = kan(hd) [N,128] ----
  for (int c = 0; c < N_NODES / CHUNK; ++c) {
    expand_kernel<<<blocks((long long)CHUNK * HDIM, 256), 256, 0, stream>>>(
        bufB + (size_t)c * CHUNK * HDIM, expb, CHUNK, HDIM);
    launch_gemm(expb, wp_feat, recon_out + (size_t)c * CHUNK * F_IN, CHUNK, F_IN, 9 * HDIM, nullptr, 0, stream);
  }

  // ---- degree decoder: degrees = relu(kan2(relu(kan1(z)))) ----
  for (int c = 0; c < N_NODES / CHUNK; ++c) {
    expand_kernel<<<blocks((long long)CHUNK * LDIM, 256), 256, 0, stream>>>(
        z + (size_t)c * CHUNK * LDIM, expb, CHUNK, LDIM);
    launch_gemm(expb, wp_dk1, ehb, CHUNK, HDIM, 9 * LDIM, nullptr, 1, stream);
    kan_out1_kernel<1><<<blocks(CHUNK, 8) , 256, 0, stream>>>(
        ehb, CHUNK, HDIM, dk2_bw, dk2_sw, dk2_sc, deg_out + (size_t)c * CHUNK);
  }

  // ---- edge decoder: sigmoid(kan2(kan1(z[p0]*z[p1]))) ----
  for (int c = 0; c < EP_PRED / CHUNK; ++c) {
    gather_mul_kernel<<<blocks((long long)CHUNK * LDIM, 256), 256, 0, stream>>>(
        p0, p1, z, efb, c * CHUNK, CHUNK);
    expand_kernel<<<blocks((long long)CHUNK * LDIM, 256), 256, 0, stream>>>(
        efb, expb, CHUNK, LDIM);
    launch_gemm(expb, wp_ek1, ehb, CHUNK, HDIM, 9 * LDIM, nullptr, 0, stream);
    kan_out1_kernel<0><<<blocks(CHUNK, 8), 256, 0, stream>>>(
        ehb, CHUNK, HDIM, ek2_bw, ek2_sw, ek2_sc, edge_out + (size_t)c * CHUNK);
  }
}
